// AdaptiveEpisodicMemory_5153960755776
// MI455X (gfx1250) — compile-verified, matrix operand order checked by reading, not run on hardware
//
#include <hip/hip_runtime.h>

// ---------------------------------------------------------------------------
// Fused episodic-memory attention for gfx1250 (MI455X), wave32 + WMMA f16.
//   out[b,:] = softmax_m( Q[b]·K[m] + 0.5*CTX[b]·MC[m] + 0.3*exp(-0.1*(1-ts[m]))
//                         masked by used[m] ) · V[m,:]
// Two-kernel plan:
//   1) prep: one-time f32->f16 conversion of K / V(transposed) / MC + bias[M]
//      into workspace (20.25 MB, resident in the 192 MB L2 afterwards).
//   2) attn: flash-attention over M; tiles staged into LDS with
//      GLOBAL_LOAD_ASYNC_TO_LDS_B128 (ASYNCcnt) when available; 20 WMMAs per
//      64-slot chunk per wave.
// ---------------------------------------------------------------------------

typedef __attribute__((ext_vector_type(16))) _Float16 v16h;
typedef __attribute__((ext_vector_type(4)))  _Float16 v4h;
typedef __attribute__((ext_vector_type(8)))  float    v8f;
typedef __attribute__((ext_vector_type(4)))  int      v4i;

#define B_    4096
#define M_    65536
#define D_    64
#define CD_   32

#define TQ     32                 // queries per workgroup (2 tiles of 16)
#define NSEG   4                  // M splits inside a workgroup
#define BK     64                 // memory slots per chunk
#define SEGLEN (M_ / NSEG)        // 16384
#define NITER  (SEGLEN / BK)      // 256

#if defined(__has_builtin)
#if __has_builtin(__builtin_amdgcn_global_load_async_to_lds_b128)
#define ASYNC_LDS 1
#endif
#endif

// 16-byte global->LDS copy: async (no VGPR round trip, ASYNCcnt) or fallback.
__device__ __forceinline__ void cp16(void* lds_dst, const void* gsrc) {
#ifdef ASYNC_LDS
  // Builtin expects (int4 AS1*, int4 AS3*, imm offset, imm cpol).
  // AS3 pointer == LDS byte offset == low 32 bits of the flat shared address
  // (ISA 10.2 aperture layout); AS1 pointer == flat global address.
  __builtin_amdgcn_global_load_async_to_lds_b128(
      (__attribute__((address_space(1))) v4i*)(uintptr_t)gsrc,
      (__attribute__((address_space(3))) v4i*)(unsigned)(uintptr_t)lds_dst,
      0, 0);
#else
  *(uint4*)lds_dst = *(const uint4*)gsrc;
#endif
}

__device__ __forceinline__ void cp_wait() {
#ifdef ASYNC_LDS
#if __has_builtin(__builtin_amdgcn_s_wait_asynccnt)
  __builtin_amdgcn_s_wait_asynccnt(0);
#else
  asm volatile("s_wait_asynccnt 0" ::: "memory");
#endif
#endif
}

// A/B fragment k-pair base for 16-bit WMMA operands (ISA 7.12.2, 16x32 layout).
__device__ __forceinline__ int kbase(int v, int half) {
  return ((v < 4) ? (2 * v) : (2 * v + 8)) + (half << 3);
}

__device__ __forceinline__ v16h load_frag(const _Float16* row, int half, int koff) {
  v16h f;
#pragma unroll
  for (int v = 0; v < 8; ++v) {
    int kb = kbase(v, half) + koff;
    f[2 * v]     = row[kb];
    f[2 * v + 1] = row[kb + 1];
  }
  return f;
}

__device__ __forceinline__ v8f wmma16(v16h a, v16h b, v8f c) {
  return __builtin_amdgcn_wmma_f32_16x16x32_f16(false, a, false, b,
                                                (short)0, c, false, false);
}

// ---------------------------------------------------------------------------
// Kernel 1: one-time f32 -> f16 conversion + V transpose + bias precompute.
// ---------------------------------------------------------------------------
__global__ __launch_bounds__(256, 1)
void epmem_prep(const float* __restrict__ MK, const float* __restrict__ MV,
                const float* __restrict__ MC, const float* __restrict__ TS,
                const int*   __restrict__ USED,
                _Float16* __restrict__ Kh, _Float16* __restrict__ Vth,
                _Float16* __restrict__ Ch, float* __restrict__ Bias)
{
  __shared__ alignas(16) _Float16 sT[64][80];   // V^T staging (160B rows, 16B ok)
  const int tid = threadIdx.x;
  const int m0  = blockIdx.x * 64;

  const float4* K4 = (const float4*)(MK + (size_t)m0 * D_);
  const float4* V4 = (const float4*)(MV + (size_t)m0 * D_);
  const float4* C4 = (const float4*)(MC + (size_t)m0 * CD_);

#pragma unroll
  for (int e = tid; e < 64 * D_ / 4; e += 256) {       // K straight convert
    float4 k = K4[e];
    v4h h = {(_Float16)k.x, (_Float16)k.y, (_Float16)k.z, (_Float16)k.w};
    *(v4h*)(Kh + (size_t)m0 * D_ + e * 4) = h;
  }
#pragma unroll
  for (int e = tid; e < 64 * CD_ / 4; e += 256) {      // MC straight convert
    float4 c = C4[e];
    v4h h = {(_Float16)c.x, (_Float16)c.y, (_Float16)c.z, (_Float16)c.w};
    *(v4h*)(Ch + (size_t)m0 * CD_ + e * 4) = h;
  }
#pragma unroll
  for (int e = tid; e < 64 * D_ / 4; e += 256) {       // V -> LDS transposed
    int r = e >> 4, c4 = (e & 15) * 4;
    float4 v = V4[e];
    sT[c4 + 0][r] = (_Float16)v.x;
    sT[c4 + 1][r] = (_Float16)v.y;
    sT[c4 + 2][r] = (_Float16)v.z;
    sT[c4 + 3][r] = (_Float16)v.w;
  }
  if (tid < 64) {
    int s = m0 + tid;
    float dec = 0.3f * __expf(-0.1f * (1.0f - TS[s]));
    Bias[s] = USED[s] ? dec : -1e9f;
  }
  __syncthreads();
#pragma unroll
  for (int e = tid; e < 64 * D_ / 8; e += 256) {       // LDS -> Vth coalesced
    int d = e >> 3, c8 = (e & 7) * 8;
    *(uint4*)(Vth + (size_t)d * M_ + m0 + c8) = *(const uint4*)&sT[d][c8];
  }
}

// ---------------------------------------------------------------------------
// Kernel 2: flash attention.  PRE=true: stage pre-converted f16 via async LDS.
//           PRE=false: self-contained f32 path (workspace too small).
// ---------------------------------------------------------------------------
template <bool PRE>
__global__ __launch_bounds__(256, 1)
void epmem_attn(const float* __restrict__ Q,  const float* __restrict__ CTX,
                const float* __restrict__ MK, const float* __restrict__ MV,
                const float* __restrict__ MC, const float* __restrict__ TS,
                const int*   __restrict__ USED,
                const _Float16* __restrict__ Kh, const _Float16* __restrict__ Vth,
                const _Float16* __restrict__ Ch, const float* __restrict__ Bias,
                float* __restrict__ OUT)
{
  __shared__ alignas(16) _Float16 sQ [TQ][D_ + 8];
  __shared__ alignas(16) _Float16 sCq[TQ][CD_ + 8];
  __shared__ alignas(16) _Float16 sK [NSEG][BK][D_ + 8];   // [slot][dim], 144B rows
  __shared__ alignas(16) _Float16 sVt[NSEG][D_][BK + 8];   // [dim][slot]
  __shared__ alignas(16) _Float16 sC [NSEG][BK][CD_ + 8];  // 80B rows
  __shared__ alignas(16) float    sBias[NSEG][BK];
  __shared__ alignas(16) _Float16 sP [8][16][D_ + 8];
  __shared__ float    sStat[2][NSEG][16][2];
  __shared__ float    sO [TQ][D_];

  const int tid   = threadIdx.x;
  const int lane  = tid & 31;
  const int wave  = tid >> 5;
  const int qtile = wave & 1;
  const int seg   = wave >> 1;
  const int half  = lane >> 4;
  const int l16   = lane & 15;
  const int qrow0 = blockIdx.x * TQ;

  for (int e = tid; e < TQ * D_; e += 256) sO[e >> 6][e & 63] = 0.f;
  for (int e = tid; e < TQ * D_; e += 256) {
    int r = e >> 6, c = e & 63;
    sQ[r][c] = (_Float16)Q[(qrow0 + r) * D_ + c];
  }
  for (int e = tid; e < TQ * CD_; e += 256) {
    int r = e >> 5, c = e & 31;
    sCq[r][c] = (_Float16)(0.5f * CTX[(qrow0 + r) * CD_ + c]);  // fold the 0.5
  }
  __syncthreads();

  const _Float16* qrow = &sQ[qtile * 16 + l16][0];
  v16h aQ0 = load_frag(qrow, half, 0);
  v16h aQ1 = load_frag(qrow, half, 32);
  v16h aC  = load_frag(&sCq[qtile * 16 + l16][0], half, 0);

  float mrow[8], lsum[8];
  v8f o[4] = {};
#pragma unroll
  for (int r = 0; r < 8; ++r) { mrow[r] = -1e30f; lsum[r] = 0.f; }

  const int stid    = (wave & 1) * 32 + lane;   // 0..63 across the wave pair
  const int segbase = seg * SEGLEN;

  for (int it = 0; it < NITER; ++it) {
    const int mstart = segbase + it * BK;
    __syncthreads();   // previous chunk fully consumed

    if (PRE) {
      // ---- pure f16 tile copies, async global->LDS when available ----
      const _Float16* gK = Kh  + (size_t)mstart * D_;
      const _Float16* gV = Vth + mstart;            // + d*M_ per dim row
      const _Float16* gC = Ch  + (size_t)mstart * CD_;
#pragma unroll
      for (int e = stid; e < BK * D_ / 8; e += 64) {   // K and V^T, 8 cps each
        int r = e >> 3, c8 = (e & 7) * 8;
        cp16(&sK [seg][r][c8], gK + r * D_ + c8);
        cp16(&sVt[seg][r][c8], gV + (size_t)r * M_ + c8);
      }
#pragma unroll
      for (int e = stid; e < BK * CD_ / 8; e += 64) {  // contexts
        int r = e >> 2, c8 = (e & 3) * 8;
        cp16(&sC[seg][r][c8], gC + r * CD_ + c8);
      }
      if (stid < BK / 4) cp16(&sBias[seg][stid * 4], Bias + mstart + stid * 4);
      cp_wait();
    } else {
      // ---- fallback: convert f32 inputs in-kernel ----
      const float4* K4 = (const float4*)(MK + (size_t)mstart * D_);
      const float4* V4 = (const float4*)(MV + (size_t)mstart * D_);
      const float4* C4 = (const float4*)(MC + (size_t)mstart * CD_);
#pragma unroll 4
      for (int e = stid; e < BK * D_ / 4; e += 64) {
        int r = e >> 4, c4 = (e & 15) * 4;
        float4 kk = K4[e];
        sK[seg][r][c4 + 0] = (_Float16)kk.x;
        sK[seg][r][c4 + 1] = (_Float16)kk.y;
        sK[seg][r][c4 + 2] = (_Float16)kk.z;
        sK[seg][r][c4 + 3] = (_Float16)kk.w;
        float4 vv = V4[e];
        sVt[seg][c4 + 0][r] = (_Float16)vv.x;
        sVt[seg][c4 + 1][r] = (_Float16)vv.y;
        sVt[seg][c4 + 2][r] = (_Float16)vv.z;
        sVt[seg][c4 + 3][r] = (_Float16)vv.w;
      }
#pragma unroll 4
      for (int e = stid; e < BK * CD_ / 4; e += 64) {
        int r = e >> 3, c4 = (e & 7) * 4;
        float4 cc = C4[e];
        sC[seg][r][c4 + 0] = (_Float16)cc.x;
        sC[seg][r][c4 + 1] = (_Float16)cc.y;
        sC[seg][r][c4 + 2] = (_Float16)cc.z;
        sC[seg][r][c4 + 3] = (_Float16)cc.w;
      }
      {
        int s = mstart + stid;
        float dec = 0.3f * __expf(-0.1f * (1.0f - TS[s]));
        sBias[seg][stid] = USED[s] ? dec : -1e9f;
      }
      if (it + 1 < NITER) {   // global_prefetch_b8 for next chunk
        __builtin_prefetch(MK + (size_t)(mstart + BK) * D_ + stid * 64, 0, 1);
        __builtin_prefetch(MV + (size_t)(mstart + BK) * D_ + stid * 64, 0, 1);
      }
    }
    __syncthreads();

    // ---- S = Q K^T + 0.5 CTX MC^T + bias : four 16x16 tiles ----
    v8f s[4];
#pragma unroll
    for (int nt = 0; nt < 4; ++nt) {
      int n = nt * 16 + l16;
      v8f acc = {};
      acc = wmma16(aQ0, load_frag(&sK[seg][n][0], half, 0),  acc);
      acc = wmma16(aQ1, load_frag(&sK[seg][n][0], half, 32), acc);
      acc = wmma16(aC,  load_frag(&sC[seg][n][0], half, 0),  acc);
      float bias = sBias[seg][n];
#pragma unroll
      for (int r = 0; r < 8; ++r) acc[r] += bias;
      s[nt] = acc;
    }

    // ---- online softmax (rows live across 16 lanes of each half) ----
    float mnew[8], scl[8];
#pragma unroll
    for (int r = 0; r < 8; ++r) {
      float cm = fmaxf(fmaxf(s[0][r], s[1][r]), fmaxf(s[2][r], s[3][r]));
#pragma unroll
      for (int off = 1; off < 16; off <<= 1)
        cm = fmaxf(cm, __shfl_xor(cm, off, 32));
      mnew[r] = fmaxf(mrow[r], cm);
      scl[r]  = __expf(mrow[r] - mnew[r]);
      lsum[r] *= scl[r];
      mrow[r]  = mnew[r];
    }
#pragma unroll
    for (int dt = 0; dt < 4; ++dt)
#pragma unroll
      for (int r = 0; r < 8; ++r) o[dt][r] *= scl[r];

#pragma unroll
    for (int nt = 0; nt < 4; ++nt) {
      int n = nt * 16 + l16;
#pragma unroll
      for (int r = 0; r < 8; ++r) {
        float p = __expf(s[nt][r] - mnew[r]);
        lsum[r] += p;
        sP[wave][half * 8 + r][n] = (_Float16)p;   // C-layout -> LDS
      }
    }
    // wave-private LDS region; DS ops from one wave complete in order.

    // ---- O += P V : P re-read in A layout, V^T gives B fragments ----
    v16h p0 = load_frag(&sP[wave][l16][0], half, 0);
    v16h p1 = load_frag(&sP[wave][l16][0], half, 32);
#pragma unroll
    for (int dt = 0; dt < 4; ++dt) {
      int n = dt * 16 + l16;
      v16h b0 = load_frag(&sVt[seg][n][0], half, 0);
      v16h b1 = load_frag(&sVt[seg][n][0], half, 32);
      o[dt] = wmma16(p0, b0, o[dt]);
      o[dt] = wmma16(p1, b1, o[dt]);
    }
  }

  // ---- merge the 4 M-segments per query tile ----
#pragma unroll
  for (int r = 0; r < 8; ++r) {
#pragma unroll
    for (int off = 1; off < 16; off <<= 1)
      lsum[r] += __shfl_xor(lsum[r], off, 32);
  }
  if (l16 == 0) {
#pragma unroll
    for (int r = 0; r < 8; ++r) {
      sStat[qtile][seg][half * 8 + r][0] = mrow[r];
      sStat[qtile][seg][half * 8 + r][1] = lsum[r];
    }
  }
  __syncthreads();

#pragma unroll
  for (int r = 0; r < 8; ++r) {
    int row = half * 8 + r;
    float gm = -1e30f;
    for (int s2 = 0; s2 < NSEG; ++s2) gm = fmaxf(gm, sStat[qtile][s2][row][0]);
    float gl = 0.f;
    for (int s2 = 0; s2 < NSEG; ++s2)
      gl += sStat[qtile][s2][row][1] * __expf(sStat[qtile][s2][row][0] - gm);
    float msc = __expf(mrow[r] - gm) / gl;
#pragma unroll
    for (int dt = 0; dt < 4; ++dt)
      atomicAdd(&sO[qtile * 16 + row][dt * 16 + l16], o[dt][r] * msc); // ds_add_f32
  }
  __syncthreads();

  for (int e = tid; e < TQ * D_; e += 256) {
    int r = e >> 6, c = e & 63;
    OUT[(size_t)(qrow0 + r) * D_ + c] = sO[r][c];
  }
}

extern "C" void kernel_launch(void* const* d_in, const int* in_sizes, int n_in,
                              void* d_out, int out_size, void* d_ws, size_t ws_size,
                              hipStream_t stream) {
  const float* Q   = (const float*)d_in[0];  // query          [B, D]
  const float* CTX = (const float*)d_in[1];  // context        [B, CD]
  const float* MK  = (const float*)d_in[2];  // mem_keys       [M, D]
  const float* MV  = (const float*)d_in[3];  // mem_values     [M, D]
  const float* MC  = (const float*)d_in[4];  // mem_contexts   [M, CD]
  const float* TS  = (const float*)d_in[5];  // mem_timestamps [M]
  const int*   US  = (const int*)d_in[6];    // used_slots     [M]
  (void)in_sizes; (void)n_in; (void)out_size;

  const size_t szK = (size_t)M_ * D_  * sizeof(_Float16);  // 8 MB
  const size_t szV = (size_t)M_ * D_  * sizeof(_Float16);  // 8 MB
  const size_t szC = (size_t)M_ * CD_ * sizeof(_Float16);  // 4 MB
  const size_t szB = (size_t)M_ * sizeof(float);           // 256 KB
  const size_t need = szK + szV + szC + szB;

  if (d_ws && ws_size >= need) {
    char* w = (char*)d_ws;
    _Float16* Kh  = (_Float16*)(w);
    _Float16* Vth = (_Float16*)(w + szK);
    _Float16* Ch  = (_Float16*)(w + szK + szV);
    float*    Bs  = (float*)   (w + szK + szV + szC);
    epmem_prep<<<M_ / 64, 256, 0, stream>>>(MK, MV, MC, TS, US, Kh, Vth, Ch, Bs);
    epmem_attn<true><<<B_ / TQ, 256, 0, stream>>>(Q, CTX, MK, MV, MC, TS, US,
                                                  Kh, Vth, Ch, Bs, (float*)d_out);
  } else {
    epmem_attn<false><<<B_ / TQ, 256, 0, stream>>>(Q, CTX, MK, MV, MC, TS, US,
                                                   nullptr, nullptr, nullptr,
                                                   nullptr, (float*)d_out);
  }
}
